// LocalSparseAttention_85066122265495
// MI455X (gfx1250) — matrix operand order repaired and checked by reading, not compiled, not run on hardware
//
#include <hip/hip_runtime.h>

// ---------------------------------------------------------------------------
// LocalSparseAttention on MI455X (gfx1250, wave32, WMMA bf16, TDM staging)
// B=8, S=4096, H=1024, WIN=128, TOPK=32
// ---------------------------------------------------------------------------

#define B_    8
#define S_    4096
#define H_    1024
#define WIN_  128
#define TOPK_ 32

typedef __bf16 bf16x16 __attribute__((ext_vector_type(16)));
typedef float  f32x8   __attribute__((ext_vector_type(8)));
typedef unsigned int u32x4 __attribute__((ext_vector_type(4)));
typedef int          i32x4 __attribute__((ext_vector_type(4)));
typedef int          i32x8 __attribute__((ext_vector_type(8)));

union BF16Frag { bf16x16 v; uint4 q[2]; };
union U4U16    { uint4 u; unsigned short s[8]; };

#if defined(__has_builtin)
#if __has_builtin(__builtin_amdgcn_tensor_load_to_lds)
#define HAVE_TDM 1
#endif
#endif
#ifndef HAVE_TDM
#define HAVE_TDM 0
#endif

__device__ __forceinline__ unsigned short f2bf(float f) {
  unsigned u = __float_as_uint(f);
  u += 0x7FFFu + ((u >> 16) & 1u);          // round-to-nearest-even
  return (unsigned short)(u >> 16);
}
__device__ __forceinline__ unsigned pack2bf(float lo, float hi) {
  return (unsigned)f2bf(lo) | ((unsigned)f2bf(hi) << 16);
}

// order-preserving f32-bits -> u32 map (for top-k threshold search)
__device__ __forceinline__ unsigned ordbits(unsigned u) {
  return (u & 0x80000000u) ? ~u : (u | 0x80000000u);
}
__device__ __forceinline__ unsigned ordkey(float f) { return ordbits(__float_as_uint(f)); }
__device__ __forceinline__ float ordinv(unsigned k) {
  return __uint_as_float((k & 0x80000000u) ? (k & 0x7FFFFFFFu) : ~k);
}

// A fragment (16xK=32 bf16) per CDNA5 ISA layout:
//   lane<16 : M=lane,    K = {0..7, 16..23}
//   lane>=16: M=lane-16, K = {8..15, 24..31}
__device__ __forceinline__ bf16x16 load_a_frag(const unsigned short* p, int stride,
                                               int m0, int kbase, int lane) {
  const int half = lane >> 4;
  const unsigned short* rp = p + (m0 + (lane & 15)) * stride + kbase + half * 8;
  BF16Frag u;
  u.q[0] = *(const uint4*)(rp);
  u.q[1] = *(const uint4*)(rp + 16);
  return u.v;
}

// B fragment (K=32 x N=16 bf16): lanes 0-15 hold K={0..7, 16..23} of column N=lane,
// lanes 16-31 hold K={8..15, 24..31}.  Source tile stored [N][K] row-major in LDS.
__device__ __forceinline__ bf16x16 load_b_frag(const unsigned short* p, int stride,
                                               int n0, int kbase, int lane) {
  const int half = lane >> 4;
  const unsigned short* rp = p + (n0 + (lane & 15)) * stride + kbase + half * 8;
  BF16Frag u;
  u.q[0] = *(const uint4*)(rp);
  u.q[1] = *(const uint4*)(rp + 16);
  return u.v;
}

#if HAVE_TDM
// TDM: DMA a 2D tile (tile_w elems x tile_h rows, 2-byte elems) from global to
// LDS.  D# bit packing per CDNA5 ISA ch.8 (group0: 128b, group1: 256b).
__device__ __forceinline__ void tdm_load_2d_bf16(
    unsigned lds_byte_off, const void* gaddr,
    unsigned tensor_w, unsigned tensor_h,
    unsigned tile_w, unsigned tile_h, unsigned row_stride_elems) {
  const unsigned long long ga = (unsigned long long)gaddr;
  u32x4 g0;
  g0[0] = 1u;                                   // count=1 valid user descriptor
  g0[1] = lds_byte_off;                         // lds_addr (bytes)
  g0[2] = (unsigned)ga;                         // global_addr[31:0]
  g0[3] = ((unsigned)(ga >> 32) & 0x01FFFFFFu)  // global_addr[56:32]
          | (2u << 30);                         // type = 2 ("image")
  i32x8 g1;
  g1[0] = (int)(1u << 16);                      // wg_mask=0, data_size=1 (2B)
  g1[1] = (int)(tensor_w << 16);                // tensor_dim0[15:0] @ [63:48]
  g1[2] = (int)((tensor_w >> 16) | (tensor_h << 16));   // dim0 hi, dim1 lo
  g1[3] = (int)((tensor_h >> 16) | (tile_w << 16));     // dim1 hi, tile_dim0
  g1[4] = (int)tile_h;                          // tile_dim1 (tile_dim2 = 0)
  g1[5] = (int)row_stride_elems;                // tensor_dim0_stride[31:0]
  g1[6] = 0;                                    // stride hi, dim1_stride lo
  g1[7] = 0;
  const i32x4 z4 = {0, 0, 0, 0};
#if __clang_major__ >= 23
  const i32x8 z8 = {0, 0, 0, 0, 0, 0, 0, 0};
  __builtin_amdgcn_tensor_load_to_lds(g0, g1, z4, z4, z8, 0);
#else
  __builtin_amdgcn_tensor_load_to_lds(g0, g1, z4, z4, 0);
#endif
}
#endif

// ---------------------------------------------------------------------------
// Kernel 1: q/k/v = x @ W^T + b   (f32 in -> bf16 out), 128x128 tiles
// grid = (BS/128, 24): y selects {q,k,v} x (1024/128) column blocks
// ---------------------------------------------------------------------------
__global__ __launch_bounds__(256) void qkv_proj_kernel(
    const float* __restrict__ x,
    const float* __restrict__ Wq, const float* __restrict__ bq,
    const float* __restrict__ Wk, const float* __restrict__ bk,
    const float* __restrict__ Wv, const float* __restrict__ bv,
    unsigned short* __restrict__ qw, unsigned short* __restrict__ kw,
    unsigned short* __restrict__ vw) {
  __shared__ unsigned short xa[128 * 32];   // A tile: 128 rows x 32 K (bf16)
  __shared__ unsigned short wb[128 * 32];   // B tile: 128 cols x 32 K (bf16)

  const int tid  = threadIdx.x;
  const int wave = tid >> 5, lane = tid & 31;
  const int m0   = blockIdx.x * 128;
  const int nb   = blockIdx.y;
  const int mat  = nb >> 3;
  const int cb   = (nb & 7) * 128;
  const float* W    = (mat == 0) ? Wq : (mat == 1) ? Wk : Wv;
  const float* bias = (mat == 0) ? bq : (mat == 1) ? bk : bv;
  unsigned short* out = (mat == 0) ? qw : (mat == 1) ? kw : vw;

  f32x8 acc[8];
#pragma unroll
  for (int t = 0; t < 8; ++t)
#pragma unroll
    for (int e = 0; e < 8; ++e) acc[t][e] = 0.0f;

  const int sr = tid >> 1, sc = (tid & 1) * 16;   // staging: 2 threads per row

  for (int kk = 0; kk < H_ / 32; ++kk) {
    const float* xs = x + (size_t)(m0 + sr) * H_ + kk * 32 + sc;
    const float* ws = W + (size_t)(cb + sr) * H_ + kk * 32 + sc;
    if (kk + 1 < H_ / 32) {                       // global_prefetch_b8
      __builtin_prefetch(xs + 32, 0, 1);
      __builtin_prefetch(ws + 32, 0, 1);
    }
#pragma unroll
    for (int i = 0; i < 16; i += 4) {             // packed bf16x2 LDS stores
      float4 xv = *(const float4*)(xs + i);
      float4 wv = *(const float4*)(ws + i);
      *(unsigned*)(xa + sr * 32 + sc + i)     = pack2bf(xv.x, xv.y);
      *(unsigned*)(xa + sr * 32 + sc + i + 2) = pack2bf(xv.z, xv.w);
      *(unsigned*)(wb + sr * 32 + sc + i)     = pack2bf(wv.x, wv.y);
      *(unsigned*)(wb + sr * 32 + sc + i + 2) = pack2bf(wv.z, wv.w);
    }
    __syncthreads();

    bf16x16 a = load_a_frag(xa, 32, wave * 16, 0, lane);
#pragma unroll
    for (int t = 0; t < 8; ++t) {
      bf16x16 b = load_b_frag(wb, 32, t * 16, 0, lane);
      acc[t] = __builtin_amdgcn_wmma_f32_16x16x32_bf16(
          false, a, false, b, (short)0, acc[t], false, false);
    }
    __syncthreads();
  }

  // epilogue: + bias, f32 -> bf16, store.  C layout: VGPR a -> M = a + 8*half
  const int half = lane >> 4, n15 = lane & 15;
#pragma unroll
  for (int t = 0; t < 8; ++t) {
    const int col = cb + t * 16 + n15;
    const float bcol = bias[col];
#pragma unroll
    for (int a = 0; a < 8; ++a) {
      const int m = m0 + wave * 16 + half * 8 + a;
      out[(size_t)m * H_ + col] = f2bf(acc[t][a] + bcol);
    }
  }
}

// ---------------------------------------------------------------------------
// Kernel 2: per-window attention. grid = B * (S/WIN) = 256 blocks.
//   scores = QK^T/32 (WMMA, q/k tiles DMA'd by the TDM) -> top-32 mask
//   -> softmax -> P@V (WMMA) -> LeakyReLU + residual -> d_out (pre-LN)
// Dynamic LDS (128 KB):
//   [0,64K)     scores f32 [128][128]
//   [64K,128K)  okeys u32 [128][128]   (phase B only; overlays the two below)
//   [64K,96K)   probs bf16 [128][128]  (written after okeys are dead)
//   [96K,128K)  phase A: qtile(8K)+ktile(8K) | phase C: v^T bf16 [128][128]
// ---------------------------------------------------------------------------
__global__ __launch_bounds__(256) void attn_kernel(
    const float* __restrict__ x,
    const unsigned short* __restrict__ qw,
    const unsigned short* __restrict__ kw,
    const unsigned short* __restrict__ vw,
    float* __restrict__ out) {
  extern __shared__ unsigned char smem[];
  float*          scores = (float*)smem;
  unsigned*       okeys  = (unsigned*)(smem + 65536);
  unsigned short* probs  = (unsigned short*)(smem + 65536);
  unsigned short* qtl    = (unsigned short*)(smem + 65536 + 32768);
  unsigned short* ktl    = qtl + 128 * 32;
  unsigned short* vt     = qtl;                 // overlays q/k tiles in phase C

  const int tid  = threadIdx.x;
  const int wave = tid >> 5, lane = tid & 31;
  const int half = lane >> 4, n15 = lane & 15;
  const int row0 = blockIdx.x * 128;            // windows tile flat B*S rows

  // ---- Phase A: scores = Q K^T / sqrt(H) -----------------------------------
  f32x8 sacc[8];
#pragma unroll
  for (int t = 0; t < 8; ++t)
#pragma unroll
    for (int e = 0; e < 8; ++e) sacc[t][e] = 0.0f;

#if HAVE_TDM
  const unsigned qtl_off = (unsigned)(unsigned long long)(const void*)qtl;
  const unsigned ktl_off = (unsigned)(unsigned long long)(const void*)ktl;
#endif

  for (int kk = 0; kk < H_ / 32; ++kk) {
#if HAVE_TDM
    // Tensor Data Mover: one wave DMAs each 128x32 bf16 tile into LDS.
    if (wave == 0) {
      tdm_load_2d_bf16(qtl_off, qw + (size_t)row0 * H_ + kk * 32,
                       H_, (unsigned)(B_ * S_), 32, 128, H_);
      __builtin_amdgcn_s_wait_tensorcnt(0);
    } else if (wave == 1) {
      tdm_load_2d_bf16(ktl_off, kw + (size_t)row0 * H_ + kk * 32,
                       H_, (unsigned)(B_ * S_), 32, 128, H_);
      __builtin_amdgcn_s_wait_tensorcnt(0);
    }
#else
    {
      const int sr = tid >> 1, sc = (tid & 1) * 16;
      const uint4* qs = (const uint4*)(qw + (size_t)(row0 + sr) * H_ + kk * 32 + sc);
      const uint4* ks = (const uint4*)(kw + (size_t)(row0 + sr) * H_ + kk * 32 + sc);
      *(uint4*)(qtl + sr * 32 + sc)     = qs[0];
      *(uint4*)(qtl + sr * 32 + sc + 8) = qs[1];
      *(uint4*)(ktl + sr * 32 + sc)     = ks[0];
      *(uint4*)(ktl + sr * 32 + sc + 8) = ks[1];
    }
#endif
    __syncthreads();

    bf16x16 a = load_a_frag(qtl, 32, wave * 16, 0, lane);
#pragma unroll
    for (int t = 0; t < 8; ++t) {
      bf16x16 b = load_b_frag(ktl, 32, t * 16, 0, lane);
      sacc[t] = __builtin_amdgcn_wmma_f32_16x16x32_bf16(
          false, a, false, b, (short)0, sacc[t], false, false);
    }
    __syncthreads();
  }
#pragma unroll
  for (int t = 0; t < 8; ++t)
#pragma unroll
    for (int a = 0; a < 8; ++a)
      scores[(wave * 16 + half * 8 + a) * 128 + t * 16 + n15] =
          sacc[t][a] * 0.03125f;               // 1/sqrt(1024)
  __syncthreads();

  // ---- Phase B0: order-preserving keys (vectorized, all threads) -----------
  {
    const uint4* s4 = (const uint4*)scores + tid * 16;   // 64 floats/thread
    uint4* k4 = (uint4*)okeys + tid * 16;
#pragma unroll
    for (int i = 0; i < 16; ++i) {
      uint4 s = s4[i];
      uint4 k;
      k.x = ordbits(s.x); k.y = ordbits(s.y);
      k.z = ordbits(s.z); k.w = ordbits(s.w);
      k4[i] = k;
    }
  }
  __syncthreads();

  // ---- Phase B1: per-row top-32 threshold (radix descent over keys) --------
  unsigned prefix = 0;
  float mx = 0.0f;
  if (tid < 128) {
    const uint4* krow = (const uint4*)(okeys + tid * 128);
    unsigned umax = 0;
    for (int j = 0; j < 32; ++j) {
      uint4 k = krow[j];
      umax = max(max(max(umax, k.x), max(k.y, k.z)), k.w);
    }
    for (int bit = 31; bit >= 0; --bit) {
      const unsigned cand = prefix | (1u << bit);
      int cnt = 0;
      for (int j = 0; j < 32; ++j) {
        uint4 k = krow[j];
        cnt += (k.x >= cand) + (k.y >= cand) + (k.z >= cand) + (k.w >= cand);
      }
      if (cnt >= TOPK_) prefix = cand;
    }
    mx = ordinv(umax);                        // row max = top-1 (monotone map)
  }
  __syncthreads();                            // okeys dead after this point

  // ---- Phase B2: masked softmax -> bf16 probs ------------------------------
  if (tid < 128) {
    const float* srow = scores + tid * 128;
    float denom = 0.0f;
    for (int j = 0; j < 128; ++j)
      if (ordkey(srow[j]) >= prefix) denom += __expf(srow[j] - mx);
    const float inv = 1.0f / denom;
    for (int j = 0; j < 128; j += 2) {
      const float p0 = (ordkey(srow[j]) >= prefix) ? __expf(srow[j] - mx) * inv : 0.0f;
      const float p1 = (ordkey(srow[j + 1]) >= prefix) ? __expf(srow[j + 1] - mx) * inv : 0.0f;
      *(unsigned*)(probs + tid * 128 + j) = pack2bf(p0, p1);
    }
  }
  __syncthreads();

  // ---- Phase C: out = P @ V, LeakyReLU + residual --------------------------
  for (int nc = 0; nc < 8; ++nc) {
    {   // stage V chunk transposed: vt[h][j] = v[row0+j][nc*128+h]
      const int j = tid >> 1, h0 = (tid & 1) * 64;
      const uint4* vs = (const uint4*)(vw + (size_t)(row0 + j) * H_ + nc * 128 + h0);
#pragma unroll
      for (int i = 0; i < 8; ++i) {
        U4U16 t4; t4.u = vs[i];
#pragma unroll
        for (int e = 0; e < 8; ++e) vt[(h0 + i * 8 + e) * 128 + j] = t4.s[e];
      }
    }
    __syncthreads();

    f32x8 oacc[8];
#pragma unroll
    for (int t = 0; t < 8; ++t)
#pragma unroll
      for (int e = 0; e < 8; ++e) oacc[t][e] = 0.0f;

#pragma unroll
    for (int kkk = 0; kkk < 4; ++kkk) {
      bf16x16 a = load_a_frag(probs, 128, wave * 16, kkk * 32, lane);
#pragma unroll
      for (int t = 0; t < 8; ++t) {
        bf16x16 b = load_b_frag(vt, 128, t * 16, kkk * 32, lane);
        oacc[t] = __builtin_amdgcn_wmma_f32_16x16x32_bf16(
            false, a, false, b, (short)0, oacc[t], false, false);
      }
    }

#pragma unroll
    for (int t = 0; t < 8; ++t) {
      const int col = nc * 128 + t * 16 + n15;
#pragma unroll
      for (int a = 0; a < 8; ++a) {
        const int m = wave * 16 + half * 8 + a;
        const size_t gi = (size_t)(row0 + m) * H_ + col;
        float o = oacc[t][a];
        o = (o >= 0.0f) ? o : 0.01f * o;      // LeakyReLU
        out[gi] = o + x[gi];                  // residual (pre-LayerNorm)
      }
    }
    __syncthreads();                           // before re-staging vt
  }
}

// ---------------------------------------------------------------------------
// Kernel 3: in-place LayerNorm over H=1024, one wave32 per row
// ---------------------------------------------------------------------------
__global__ __launch_bounds__(256) void layernorm_kernel(
    float* __restrict__ out, const float* __restrict__ gamma,
    const float* __restrict__ beta) {
  const int tid = threadIdx.x, wave = tid >> 5, lane = tid & 31;
  const size_t row = (size_t)blockIdx.x * 8 + wave;
  float* rp = out + row * H_;

  float4 v[8];
  float sum = 0.0f, ssq = 0.0f;
#pragma unroll
  for (int i = 0; i < 8; ++i) {
    v[i] = *(const float4*)(rp + i * 128 + lane * 4);
    sum += v[i].x + v[i].y + v[i].z + v[i].w;
    ssq += v[i].x * v[i].x + v[i].y * v[i].y + v[i].z * v[i].z + v[i].w * v[i].w;
  }
#pragma unroll
  for (int off = 16; off; off >>= 1) {
    sum += __shfl_xor(sum, off, 32);
    ssq += __shfl_xor(ssq, off, 32);
  }
  const float mu  = sum * (1.0f / (float)H_);
  const float var = ssq * (1.0f / (float)H_) - mu * mu;
  const float rs  = rsqrtf(var + 1e-5f);
#pragma unroll
  for (int i = 0; i < 8; ++i) {
    const int c = i * 128 + lane * 4;
    float4 g = *(const float4*)(gamma + c);
    float4 bb = *(const float4*)(beta + c);
    float4 o;
    o.x = (v[i].x - mu) * rs * g.x + bb.x;
    o.y = (v[i].y - mu) * rs * g.y + bb.y;
    o.z = (v[i].z - mu) * rs * g.z + bb.z;
    o.w = (v[i].w - mu) * rs * g.w + bb.w;
    *(float4*)(rp + c) = o;
  }
}

// ---------------------------------------------------------------------------
extern "C" void kernel_launch(void* const* d_in, const int* in_sizes, int n_in,
                              void* d_out, int out_size, void* d_ws, size_t ws_size,
                              hipStream_t stream) {
  const float* x     = (const float*)d_in[0];
  const float* Wq    = (const float*)d_in[1];
  const float* bq    = (const float*)d_in[2];
  const float* Wk    = (const float*)d_in[3];
  const float* bk    = (const float*)d_in[4];
  const float* Wv    = (const float*)d_in[5];
  const float* bv    = (const float*)d_in[6];
  const float* gamma = (const float*)d_in[7];
  const float* beta  = (const float*)d_in[8];
  float* out = (float*)d_out;

  const size_t n = (size_t)B_ * S_ * H_;               // 33,554,432 elements
  unsigned short* qw = (unsigned short*)d_ws;          // bf16 q  (64 MB)
  unsigned short* kw = qw + n;                         // bf16 k  (64 MB)
  unsigned short* vw = kw + n;                         // bf16 v  (64 MB)

  dim3 g1((B_ * S_) / 128, 3 * (H_ / 128));            // 256 x 24
  qkv_proj_kernel<<<g1, 256, 0, stream>>>(x, Wq, bq, Wk, bk, Wv, bv, qw, kw, vw);

  const size_t smem2 = 65536 + 65536;                  // 128 KB dynamic LDS
  attn_kernel<<<B_ * (S_ / WIN_), 256, smem2, stream>>>(x, qw, kw, vw, out);

  layernorm_kernel<<<(B_ * S_) / 8, 256, 0, stream>>>(out, gamma, beta);
}
